// MultiHeadSelfAttention_77524159693248
// MI455X (gfx1250) — matrix operand order repaired and checked by reading, not compiled, not run on hardware
//
#include <hip/hip_runtime.h>
#include <stdint.h>

// Problem constants (from reference): B=2, S=2048, E=1024, H=16, D=64
constexpr int kB = 2;
constexpr int kS = 2048;
constexpr int kE = 1024;
constexpr int kH = 16;
constexpr int kD = 64;
constexpr int kM = kB * kS;      // 4096 rows for the projections

typedef __attribute__((ext_vector_type(16))) _Float16 v16h;
typedef __attribute__((ext_vector_type(8)))  _Float16 v8h;
typedef __attribute__((ext_vector_type(4)))  _Float16 v4h;
typedef __attribute__((ext_vector_type(8)))  float    v8f;

__device__ __forceinline__ v8f wmma_f16f32(v16h a, v16h b, v8f c) {
  // D(f32 16x16) = A(16x32 f16) * B(32x16 f16) + C
  return __builtin_amdgcn_wmma_f32_16x16x32_f16(false, a, false, b, (short)0, c,
                                                false, false);
}

__device__ __forceinline__ v8f vzero8() {
  v8f z = {0.f, 0.f, 0.f, 0.f, 0.f, 0.f, 0.f, 0.f};
  return z;
}

// Async VMEM->LDS 16-byte copy (gfx1250, tracked by ASYNCcnt, bypasses VGPRs).
// VDST = LDS byte address (low 32 bits of generic shared pointer), VADDR = u64.
__device__ __forceinline__ void async_copy16(const void* g, void* l) {
  uint32_t lds = (uint32_t)(uintptr_t)l;
  asm volatile("global_load_async_to_lds_b128 %0, %1, off"
               :: "v"(lds), "v"(g) : "memory");
}
__device__ __forceinline__ void wait_async0() {
  asm volatile("s_wait_asynccnt 0x0" ::: "memory");
}

// A-matrix 16x32 f16 fragment from an LDS tile stored row(M)-major, K contiguous.
// Lane L holds row L%16; K elems (L>>4)*8+{0..7} then +16.
__device__ __forceinline__ v16h frag_a(const _Float16* row0, int stride, int lane) {
  const _Float16* p = row0 + (lane & 15) * stride + ((lane >> 4) << 3);
  v8h lo = *(const v8h*)p;
  v8h hi = *(const v8h*)(p + 16);
  return __builtin_shufflevector(lo, hi, 0,1,2,3,4,5,6,7,8,9,10,11,12,13,14,15);
}

// B-matrix 32x16 f16 fragment from an LDS tile stored col(N)-major, K contiguous.
// Lane L holds column L%16; K elems (L>>4)*8+{0..7} then +16.
__device__ __forceinline__ v16h frag_b(const _Float16* col0, int stride, int lane) {
  const _Float16* p = col0 + (lane & 15) * stride + ((lane >> 4) << 3);
  v8h lo = *(const v8h*)p;
  v8h hi = *(const v8h*)(p + 16);
  return __builtin_shufflevector(lo, hi, 0,1,2,3,4,5,6,7,8,9,10,11,12,13,14,15);
}

// ---------------------------------------------------------------- f32 -> f16
__global__ __launch_bounds__(256) void cvt_f32_to_f16(const float4* __restrict__ src,
                                                      v4h* __restrict__ dst, int n4) {
  int i = blockIdx.x * 256 + threadIdx.x;
  if (i < n4) {
    float4 v = src[i];
    v4h o = {(_Float16)v.x, (_Float16)v.y, (_Float16)v.z, (_Float16)v.w};
    dst[i] = o;
  }
}

// ------------------------------------------------------- WMMA GEMM  Y = A*W^T + b
// A: [4096][1024] f16 row-major.  W: [1024][1024] f16 row-major ([N][K], so row n
// of W is column n of B = W^T).  128x128 tile per 256-thread block (8 waves);
// each wave computes 32x64 as 2x4 WMMA 16x16 tiles, K-step 32.
// K-steps are double-buffered in LDS via async direct-to-LDS copies.
template <bool OUT_F32>
__global__ __launch_bounds__(256) void gemm_wmma(const _Float16* __restrict__ A,
                                                 const _Float16* __restrict__ W,
                                                 const float* __restrict__ bias,
                                                 void* __restrict__ out) {
  constexpr int BK = 32;
  constexpr int LSTR = 40;  // 80B row stride: 16B aligned, 16-row conflict-free
  __shared__ __align__(16) _Float16 sA[2][128 * LSTR];
  __shared__ __align__(16) _Float16 sB[2][128 * LSTR];

  const int tid  = threadIdx.x;
  const int lane = tid & 31;
  const int wid  = tid >> 5;          // 0..7
  const int wm   = (wid & 3) * 32;    // wave row base in tile
  const int wn   = (wid >> 2) * 64;   // wave col base in tile
  const int m0   = blockIdx.y * 128;
  const int n0   = blockIdx.x * 128;

  v8f acc[2][4];
#pragma unroll
  for (int mi = 0; mi < 2; ++mi)
#pragma unroll
    for (int ni = 0; ni < 4; ++ni) acc[mi][ni] = vzero8();

  const int lrow = tid >> 1;          // 0..127
  const int lk   = (tid & 1) * 16;    // 0 or 16 (halfs)

  auto issue_tile = [&](int buf, int kk) {
    const _Float16* ga = A + (size_t)(m0 + lrow) * kE + kk + lk;
    async_copy16(ga,     &sA[buf][lrow * LSTR + lk]);
    async_copy16(ga + 8, &sA[buf][lrow * LSTR + lk + 8]);
    const _Float16* gw = W + (size_t)(n0 + lrow) * kE + kk + lk;
    async_copy16(gw,     &sB[buf][lrow * LSTR + lk]);
    async_copy16(gw + 8, &sB[buf][lrow * LSTR + lk + 8]);
  };

  issue_tile(0, 0);
  wait_async0();
  __syncthreads();

  for (int kk = 0; kk < kE; kk += BK) {
    const int cur = (kk >> 5) & 1;
    if (kk + BK < kE) issue_tile(cur ^ 1, kk + BK);  // prefetch next K-step

    const _Float16* cA = sA[cur];
    const _Float16* cB = sB[cur];
    v16h af0 = frag_a(&cA[(wm + 0)  * LSTR], LSTR, lane);
    v16h af1 = frag_a(&cA[(wm + 16) * LSTR], LSTR, lane);
#pragma unroll
    for (int ni = 0; ni < 4; ++ni) {
      v16h bf = frag_b(&cB[(wn + ni * 16) * LSTR], LSTR, lane);
      acc[0][ni] = wmma_f16f32(af0, bf, acc[0][ni]);
      acc[1][ni] = wmma_f16f32(af1, bf, acc[1][ni]);
    }
    wait_async0();     // prefetch landed
    __syncthreads();   // everyone done reading cur before it becomes prefetch dst
  }

  // Epilogue: C layout -> VGPR j holds row (j + 8*(lane>=16)), col = lane%16.
  const int rowh = ((lane >> 4) & 1) * 8;
#pragma unroll
  for (int ni = 0; ni < 4; ++ni) {
    const int col = n0 + wn + ni * 16 + (lane & 15);
    const float bv = bias[col];
#pragma unroll
    for (int mi = 0; mi < 2; ++mi) {
      const int rbase = m0 + wm + mi * 16 + rowh;
#pragma unroll
      for (int j = 0; j < 8; ++j) {
        float v = acc[mi][ni][j] + bv;
        if (OUT_F32)
          ((float*)out)[(size_t)(rbase + j) * kE + col] = v;
        else
          ((_Float16*)out)[(size_t)(rbase + j) * kE + col] = (_Float16)v;
      }
    }
  }
}

// ------------------------------------------------------- flash attention kernel
// grid = (S/64, B*H), block = 128 (4 waves). Each block: 64 queries of one head.
// Q staged once via async-to-LDS; K tiles double-buffered via async-to-LDS with
// prefetch overlapping the score/softmax/PV compute; V transposed through VGPRs.
__global__ __launch_bounds__(128) void attn_wmma(const _Float16* __restrict__ Q,
                                                 const _Float16* __restrict__ Kmat,
                                                 const _Float16* __restrict__ V,
                                                 const uint8_t* __restrict__ mask,
                                                 _Float16* __restrict__ O) {
  constexpr int LSTR = 72;  // 144B stride: 16B aligned, 16-row conflict-free
  __shared__ __align__(16) _Float16 q_s[64 * LSTR];
  __shared__ __align__(16) _Float16 k_s[2][64 * LSTR];
  __shared__ __align__(16) _Float16 vt_s[64 * LSTR];      // V^T: [d][sk]
  __shared__ __align__(16) _Float16 p_s[4 * 16 * LSTR];   // per-wave P staging

  const int b   = blockIdx.y >> 4;
  const int h   = blockIdx.y & 15;
  const int s0  = blockIdx.x * 64;
  const int tid = threadIdx.x, lane = tid & 31, wid = tid >> 5;
  const size_t headoff = (size_t)h * kD;
  const size_t rowbase = (size_t)b * kS;

  // Async-stage the 64x64 K tile at sk0 into k_s[buf]: 4 chunks of 16B / thread.
  auto issue_k = [&](int buf, int sk0) {
#pragma unroll
    for (int c = 0; c < 4; ++c) {
      int chunk = c * 128 + tid;
      int r = chunk >> 3, ko = (chunk & 7) * 8;
      async_copy16(Kmat + (rowbase + sk0 + r) * kE + headoff + ko,
                   &k_s[buf][r * LSTR + ko]);
    }
  };

  // Async-stage Q tile + first K tile together.
#pragma unroll
  for (int c = 0; c < 4; ++c) {
    int chunk = c * 128 + tid;
    int r = chunk >> 3, ko = (chunk & 7) * 8;
    async_copy16(Q + (rowbase + s0 + r) * kE + headoff + ko, &q_s[r * LSTR + ko]);
  }
  issue_k(0, 0);
  wait_async0();
  __syncthreads();

  // Per-wave Q fragments (rows wid*16..+15), K=64 -> two 16x32 A-frags, in VGPRs.
  v16h qf0 = frag_a(&q_s[(wid * 16) * LSTR + 0],  LSTR, lane);
  v16h qf1 = frag_a(&q_s[(wid * 16) * LSTR + 32], LSTR, lane);

  v8f o[4];
#pragma unroll
  for (int dt = 0; dt < 4; ++dt) o[dt] = vzero8();
  float mrow[8], lrow[8];
#pragma unroll
  for (int j = 0; j < 8; ++j) { mrow[j] = -__builtin_inff(); lrow[j] = 0.f; }

  constexpr int NT = kS / 64;
  for (int kt = 0; kt < NT; ++kt) {
    const int sk0 = kt * 64;
    const int cur = kt & 1;

    // V tile transposed into LDS through VGPRs (async path cannot transpose).
#pragma unroll
    for (int c = 0; c < 4; ++c) {
      int chunk = c * 128 + tid;
      int r = chunk >> 3, ko = (chunk & 7) * 8;
      v8h vv = *(const v8h*)(V + (rowbase + sk0 + r) * kE + headoff + ko);
#pragma unroll
      for (int i = 0; i < 8; ++i) vt_s[(ko + i) * LSTR + r] = vv[i];
    }
    // Prefetch next K tile into the other buffer; completion awaited at loop end.
    if (kt + 1 < NT) issue_k(cur ^ 1, sk0 + 64);
    __syncthreads();

    // Scores S = Q K^T (16 query rows x 64 keys per wave)
    const _Float16* kc = k_s[cur];
    v8f sc[4];
#pragma unroll
    for (int ni = 0; ni < 4; ++ni) {
      sc[ni] = vzero8();
      v16h bf0 = frag_b(&kc[(ni * 16) * LSTR + 0],  LSTR, lane);
      v16h bf1 = frag_b(&kc[(ni * 16) * LSTR + 32], LSTR, lane);
      sc[ni] = wmma_f16f32(qf0, bf0, sc[ni]);
      sc[ni] = wmma_f16f32(qf1, bf1, sc[ni]);
    }

    // Scale 1/sqrt(D) + key mask (column = sk0 + ni*16 + lane%16, same for all j)
    float sv[4][8];
#pragma unroll
    for (int ni = 0; ni < 4; ++ni) {
      const int kcol = sk0 + ni * 16 + (lane & 15);
      const bool mv = mask[(size_t)b * kS + kcol] != 0;
#pragma unroll
      for (int j = 0; j < 8; ++j) {
        float t = sc[ni][j] * 0.125f;
        sv[ni][j] = mv ? t : -__builtin_inff();
      }
    }

    // Online softmax: C layout => lanes {0..15}/{16..31} each hold one row per
    // VGPR j; xor-shuffles 1,2,4,8 reduce within each 16-lane half, landing the
    // running max/sum exactly row-aligned with the accumulators.
    float pv[4][8];
#pragma unroll
    for (int j = 0; j < 8; ++j) {
      float mt = fmaxf(fmaxf(sv[0][j], sv[1][j]), fmaxf(sv[2][j], sv[3][j]));
      mt = fmaxf(mt, __shfl_xor(mt, 1, 32));
      mt = fmaxf(mt, __shfl_xor(mt, 2, 32));
      mt = fmaxf(mt, __shfl_xor(mt, 4, 32));
      mt = fmaxf(mt, __shfl_xor(mt, 8, 32));
      const float newm = fmaxf(mrow[j], mt);
      const float scl = __expf(mrow[j] - newm);
      mrow[j] = newm;
      float rs = 0.f;
#pragma unroll
      for (int ni = 0; ni < 4; ++ni) {
        float p = __expf(sv[ni][j] - newm);
        pv[ni][j] = p;
        rs += p;
      }
      rs += __shfl_xor(rs, 1, 32);
      rs += __shfl_xor(rs, 2, 32);
      rs += __shfl_xor(rs, 4, 32);
      rs += __shfl_xor(rs, 8, 32);
      lrow[j] = lrow[j] * scl + rs;
#pragma unroll
      for (int dt = 0; dt < 4; ++dt) o[dt][j] *= scl;
    }

    // P (f16) through per-wave LDS region to re-layout C->A fragments.
    _Float16* pw = &p_s[wid * 16 * LSTR];
    const int prow = ((lane >> 4) & 1) * 8;
#pragma unroll
    for (int ni = 0; ni < 4; ++ni) {
      const int pc = ni * 16 + (lane & 15);
#pragma unroll
      for (int j = 0; j < 8; ++j) pw[(prow + j) * LSTR + pc] = (_Float16)pv[ni][j];
    }
    // Same-wave LDS RAW is in-order; backend inserts s_wait_dscnt.

    // O += P @ V  (A = P 16x64, B = V^T columns d)
#pragma unroll
    for (int ks = 0; ks < 2; ++ks) {
      v16h af = frag_a(&pw[ks * 32], LSTR, lane);
#pragma unroll
      for (int dt = 0; dt < 4; ++dt) {
        v16h bf = frag_b(&vt_s[(dt * 16) * LSTR + ks * 32], LSTR, lane);
        o[dt] = wmma_f16f32(af, bf, o[dt]);
      }
    }
    wait_async0();     // K prefetch landed
    __syncthreads();   // vt_s reads done before next iteration overwrites it
  }

  // Normalize and store head output at [b, s, h*64 + d] (f16 for final GEMM)
#pragma unroll
  for (int dt = 0; dt < 4; ++dt) {
    const int col = h * kD + dt * 16 + (lane & 15);
#pragma unroll
    for (int j = 0; j < 8; ++j) {
      const int srow = s0 + wid * 16 + ((lane >> 4) & 1) * 8 + j;
      float val = o[dt][j] / lrow[j];
      O[(rowbase + srow) * kE + col] = (_Float16)val;
    }
  }
}

// ---------------------------------------------------------------- host launcher
extern "C" void kernel_launch(void* const* d_in, const int* in_sizes, int n_in,
                              void* d_out, int out_size, void* d_ws, size_t ws_size,
                              hipStream_t stream) {
  (void)in_sizes; (void)n_in; (void)out_size; (void)ws_size;
  const float*   x    = (const float*)d_in[0];
  const uint8_t* mask = (const uint8_t*)d_in[1];
  const float*   Wq   = (const float*)d_in[2];
  const float*   bq   = (const float*)d_in[3];
  const float*   Wk   = (const float*)d_in[4];
  const float*   bk   = (const float*)d_in[5];
  const float*   Wv   = (const float*)d_in[6];
  const float*   bv   = (const float*)d_in[7];
  const float*   Wo   = (const float*)d_in[8];
  const float*   bo   = (const float*)d_in[9];

  char* ws = (char*)d_ws;
  size_t off = 0;
  auto carve = [&](size_t bytes) {
    void* p = ws + off;
    off += (bytes + 255) & ~(size_t)255;
    return p;
  };
  _Float16* xh  = (_Float16*)carve((size_t)kM * kE * 2);
  _Float16* wqh = (_Float16*)carve((size_t)kE * kE * 2);
  _Float16* wkh = (_Float16*)carve((size_t)kE * kE * 2);
  _Float16* wvh = (_Float16*)carve((size_t)kE * kE * 2);
  _Float16* woh = (_Float16*)carve((size_t)kE * kE * 2);
  _Float16* qh  = (_Float16*)carve((size_t)kM * kE * 2);
  _Float16* kh  = (_Float16*)carve((size_t)kM * kE * 2);
  _Float16* vh  = (_Float16*)carve((size_t)kM * kE * 2);
  _Float16* ah  = (_Float16*)carve((size_t)kM * kE * 2);

  auto cvt = [&](const float* s, _Float16* d, int n) {
    int n4 = n >> 2;
    cvt_f32_to_f16<<<dim3((n4 + 255) / 256), dim3(256), 0, stream>>>(
        (const float4*)s, (v4h*)d, n4);
  };
  cvt(x,  xh,  kM * kE);
  cvt(Wq, wqh, kE * kE);
  cvt(Wk, wkh, kE * kE);
  cvt(Wv, wvh, kE * kE);
  cvt(Wo, woh, kE * kE);

  dim3 ggrid(kE / 128, kM / 128);  // (8, 32)
  gemm_wmma<false><<<ggrid, dim3(256), 0, stream>>>(xh, wqh, bq, (void*)qh);
  gemm_wmma<false><<<ggrid, dim3(256), 0, stream>>>(xh, wkh, bk, (void*)kh);
  gemm_wmma<false><<<ggrid, dim3(256), 0, stream>>>(xh, wvh, bv, (void*)vh);

  attn_wmma<<<dim3(kS / 64, kB * kH), dim3(128), 0, stream>>>(qh, kh, vh, mask, ah);

  gemm_wmma<true><<<ggrid, dim3(256), 0, stream>>>(ah, woh, bo, d_out);
}
